// LSTM_85246510891468
// MI455X (gfx1250) — compile-verified, matrix operand order checked by reading, not run on hardware
//
#include <hip/hip_runtime.h>
#include <math.h>

// ---------------------------------------------------------------------------
// Problem constants (match reference)
// ---------------------------------------------------------------------------
static constexpr int Bn = 128;   // batch
static constexpr int Tn = 128;   // timesteps
static constexpr int Dn = 256;   // input dim
static constexpr int Hn = 1024;  // hidden dim
static constexpr int On = 1024;  // output dim
static constexpr int KCAT = Dn + Hn;  // fused K for [x_t | h] @ [Wx | Wh]^T = 1280
static constexpr int LDA = 72;        // padded LDS row stride (halves) -> conflict-free

// ---------------------------------------------------------------------------
// CDNA5 async global->LDS path (ASYNCcnt), guarded for toolchain portability
// ---------------------------------------------------------------------------
#if defined(__has_builtin)
#if __has_builtin(__builtin_amdgcn_global_load_async_to_lds_b128) && \
    __has_builtin(__builtin_amdgcn_s_wait_asynccnt)
#define USE_ASYNC_LDS 1
#endif
#endif
#ifndef USE_ASYNC_LDS
#define USE_ASYNC_LDS 0
#endif

#if USE_ASYNC_LDS
// Builtin signature (from hipcc diagnostic): first param is
// 'int __attribute__((vector_size(16))) __device__*' (AS1), LDS param is AS3.
typedef int v4i_async __attribute__((vector_size(16)));
typedef __attribute__((address_space(1))) v4i_async v4i_g;
typedef __attribute__((address_space(3))) v4i_async v4i_l;
#endif

// ---------------------------------------------------------------------------
// CDNA5 WMMA types
// ---------------------------------------------------------------------------
typedef __attribute__((ext_vector_type(16))) __bf16 v16bf;
typedef __attribute__((ext_vector_type(8)))  float  v8f;

union Frag {
  v16bf bf;
  unsigned int u32[8];
};

__device__ __forceinline__ unsigned short f2bf(float f) {
  unsigned int u = __float_as_uint(f);
  u += 0x7fffu + ((u >> 16) & 1u);          // round-to-nearest-even
  return (unsigned short)(u >> 16);
}

__device__ __forceinline__ float sigmoidf_(float x) {
  return 1.0f / (1.0f + __expf(-x));
}

// Copy 64 contiguous bytes global -> LDS.
// Async path: GLOBAL_LOAD_ASYNC_TO_LDS_B128 x4 (no VGPR round-trip, ASYNCcnt).
// INST_OFFSET applies to both global and LDS addresses (ISA async-op pseudocode).
__device__ __forceinline__ void cp64(unsigned short* dst, const unsigned short* src) {
#if USE_ASYNC_LDS
  v4i_g* g = (v4i_g*)(unsigned short*)src;
  v4i_l* l = (v4i_l*)dst;
  __builtin_amdgcn_global_load_async_to_lds_b128(g, l, 0, 0);
  __builtin_amdgcn_global_load_async_to_lds_b128(g, l, 16, 0);
  __builtin_amdgcn_global_load_async_to_lds_b128(g, l, 32, 0);
  __builtin_amdgcn_global_load_async_to_lds_b128(g, l, 48, 0);
#else
  const uint4* s4 = reinterpret_cast<const uint4*>(src);
  uint4* d4 = reinterpret_cast<uint4*>(dst);
  d4[0] = s4[0]; d4[1] = s4[1]; d4[2] = s4[2]; d4[3] = s4[3];
#endif
}

// Copy 16 contiguous bytes global -> LDS.
__device__ __forceinline__ void cp16(unsigned short* dst, const unsigned short* src) {
#if USE_ASYNC_LDS
  __builtin_amdgcn_global_load_async_to_lds_b128(
      (v4i_g*)(unsigned short*)src, (v4i_l*)dst, 0, 0);
#else
  *reinterpret_cast<uint4*>(dst) = *reinterpret_cast<const uint4*>(src);
#endif
}

// Wait for this wave's async copies to land, then workgroup barrier.
__device__ __forceinline__ void async_join() {
#if USE_ASYNC_LDS
  __builtin_amdgcn_s_wait_asynccnt(0);
#endif
  __syncthreads();
}

// Load a 16x32 bf16 fragment from an LDS tile (row-major, stride LDA halves).
// CDNA5 16-bit fragment layout (ISA 7.12.2): lanes 0-15 carry rows, lanes 16-31
// the +8 K-offset half; VGPR v holds K pair kpat(v) = (v<4 ? 2v : 2v+8).
// B fragments mirror this with N across lanes ("one row striped across lanes").
__device__ __forceinline__ v16bf frag_lds(const unsigned short* buf, int row0, int k0) {
  const int lane = threadIdx.x & 31;
  const int r    = row0 + (lane & 15);
  const int kh   = (lane >> 4) << 3;
  Frag f;
#pragma unroll
  for (int v = 0; v < 8; ++v) {
    const int k = k0 + kh + ((v < 4) ? (v << 1) : ((v << 1) + 8));
    f.u32[v] = *reinterpret_cast<const unsigned int*>(buf + r * LDA + k);
  }
  return f.bf;
}

// ---------------------------------------------------------------------------
// Conversion / init kernels
// ---------------------------------------------------------------------------
__global__ void conv_f32_bf16(const float* __restrict__ src,
                              unsigned short* __restrict__ dst, int n) {
  int i = blockIdx.x * 256 + threadIdx.x;
  if (i < n) dst[i] = f2bf(src[i]);
}

// Fused weight matrix: Wcat[g, 0:256] = Wx[g,:], Wcat[g, 256:1280] = Wh[g,:]
__global__ void build_wcat(const float* __restrict__ Wx,
                           const float* __restrict__ Wh,
                           unsigned short* __restrict__ Wcat) {
  int i = blockIdx.x * 256 + threadIdx.x;
  if (i < 4 * Hn * KCAT) {
    int g = i / KCAT, k = i % KCAT;
    float v = (k < Dn) ? Wx[g * Dn + k] : Wh[g * Hn + (k - Dn)];
    Wcat[i] = f2bf(v);
  }
}

__global__ void init_state(unsigned short* __restrict__ h0,
                           float* __restrict__ c, int n) {
  int i = blockIdx.x * 256 + threadIdx.x;
  if (i < n) { h0[i] = 0; c[i] = 0.0f; }
}

// ---------------------------------------------------------------------------
// One LSTM timestep:  gates = [x_t | h] @ Wcat^T + bx ; cell update ; h_out
// Grid: H/32 blocks. Block: 256 threads (8 wave32).
// Wave w: n-tile nt = w&1 (16 cols), m-group mg = w>>1 (rows 32*mg..32*mg+31),
// and ALL FOUR gates for that (m,n) block -> cell update stays in-register.
// Double-buffered LDS staging via async global->LDS copies overlapped with WMMA.
// ---------------------------------------------------------------------------
__global__ __launch_bounds__(256)
void lstm_step(const unsigned short* __restrict__ xb,    // [B,T,D] bf16
               const unsigned short* __restrict__ Wcat,  // [4H,KCAT] bf16
               const float* __restrict__ bx,             // [4H]
               const unsigned short* __restrict__ h_in,  // [B,H] bf16
               unsigned short* __restrict__ h_out,       // [B,H] bf16
               float* __restrict__ c,                    // [B,H] f32
               int t) {
  __shared__ unsigned short As[2][128 * LDA];  // A chunks: 128 rows x 64 halves
  __shared__ unsigned short Ws[2][128 * LDA];  // W chunks: 4 gates x 32 cols

  const int tid  = threadIdx.x;
  const int wave = tid >> 5;
  const int lane = tid & 31;
  const int hcol0 = blockIdx.x * 32;
  const int nt = wave & 1;
  const int mg = wave >> 1;

  // Per-thread staging coordinates (64B per thread per buffer).
  const int srow = tid >> 1, shalf = tid & 1;
  const int sgate = srow >> 5;
  const size_t wrowOff = (size_t)(sgate * Hn + hcol0 + (srow & 31)) * KCAT + shalf * 32;

  v8f acc[4][2];
  const v8f vzero = {};
#pragma unroll
  for (int q = 0; q < 4; ++q) { acc[q][0] = vzero; acc[q][1] = vzero; }

  constexpr int NKC = KCAT / 64;

  // Stage chunk kc into buffer buf.
  auto stage = [&](int kc, int buf) {
    const int k0 = kc * 64;
    const unsigned short* asrc =
        (k0 < Dn) ? xb + ((size_t)srow * Tn + t) * Dn + k0 + shalf * 32
                  : h_in + (size_t)srow * Hn + (k0 - Dn) + shalf * 32;
    cp64(&As[buf][srow * LDA + shalf * 32], asrc);
    const unsigned short* wsrc = Wcat + wrowOff + k0;
    cp64(&Ws[buf][srow * LDA + shalf * 32], wsrc);
    if (kc + 2 < NKC) __builtin_prefetch(wsrc + 128, 0, 1);  // global_prefetch_b8 (L2 hint)
  };

  stage(0, 0);
  async_join();

  for (int kc = 0; kc < NKC; ++kc) {
    const int cur = kc & 1;
    if (kc + 1 < NKC) stage(kc + 1, cur ^ 1);  // overlap next chunk with compute

#pragma unroll
    for (int ks = 0; ks < 2; ++ks) {
      const int kk = ks * 32;
      v16bf a0 = frag_lds(As[cur], 32 * mg, kk);
      v16bf a1 = frag_lds(As[cur], 32 * mg + 16, kk);
#pragma unroll
      for (int q = 0; q < 4; ++q) {
        v16bf b = frag_lds(Ws[cur], q * 32 + nt * 16, kk);
        acc[q][0] = __builtin_amdgcn_wmma_f32_16x16x32_bf16(
            false, a0, false, b, (short)0, acc[q][0], false, false);
        acc[q][1] = __builtin_amdgcn_wmma_f32_16x16x32_bf16(
            false, a1, false, b, (short)0, acc[q][1], false, false);
      }
    }
    async_join();  // next buffer landed + everyone done reading cur
  }

  // Cell update, fully in-register. C/D layout: VGPR r -> M = base + r + 8*(lane>>4), N = lane&15.
  const int n = hcol0 + nt * 16 + (lane & 15);
  const float bg = bx[0 * Hn + n], bi = bx[1 * Hn + n];
  const float bff = bx[2 * Hn + n], bo = bx[3 * Hn + n];
  const int hiOff = (lane >> 4) * 8;
#pragma unroll
  for (int mt = 0; mt < 2; ++mt) {
#pragma unroll
    for (int r = 0; r < 8; ++r) {
      const int m = 32 * mg + 16 * mt + hiOff + r;
      const size_t idx = (size_t)m * Hn + n;
      const float g = tanhf(acc[0][mt][r] + bg);
      const float i = sigmoidf_(acc[1][mt][r] + bi);
      const float f = sigmoidf_(acc[2][mt][r] + bff);
      const float o = sigmoidf_(acc[3][mt][r] + bo);
      const float cn = g * i + c[idx] * f;
      c[idx] = cn;
      h_out[idx] = f2bf(tanhf(cn) * o);
    }
  }
}

// ---------------------------------------------------------------------------
// Projection: p = h_last @ Wp^T + bp  (128x1024 x 1024x1024), WMMA bf16
// ---------------------------------------------------------------------------
__global__ __launch_bounds__(256)
void proj_kernel(const unsigned short* __restrict__ hb,   // [B,H] bf16
                 const unsigned short* __restrict__ Wpb,  // [O,H] bf16
                 const float* __restrict__ bp,            // [O]
                 float* __restrict__ p) {                 // [B,O] f32
  __shared__ unsigned short As[2][128 * LDA];
  __shared__ unsigned short Ws[2][32 * LDA];

  const int tid = threadIdx.x, wave = tid >> 5, lane = tid & 31;
  const int ocol0 = blockIdx.x * 32;
  const int nt = wave & 1, mg = wave >> 1;

  v8f acc0 = {}, acc1 = {};

  const int arow = tid >> 1, ahalf = tid & 1;
  const int wrow = tid >> 3, wpart = tid & 7;  // 32 rows x (8 x 8 halves)
  constexpr int NKC = Hn / 64;

  auto stage = [&](int kc, int buf) {
    const int k0 = kc * 64;
    cp64(&As[buf][arow * LDA + ahalf * 32], hb + (size_t)arow * Hn + k0 + ahalf * 32);
    cp16(&Ws[buf][wrow * LDA + wpart * 8],
         Wpb + (size_t)(ocol0 + wrow) * Hn + k0 + wpart * 8);
  };

  stage(0, 0);
  async_join();

  for (int kc = 0; kc < NKC; ++kc) {
    const int cur = kc & 1;
    if (kc + 1 < NKC) stage(kc + 1, cur ^ 1);

#pragma unroll
    for (int ks = 0; ks < 2; ++ks) {
      const int kk = ks * 32;
      v16bf a0 = frag_lds(As[cur], 32 * mg, kk);
      v16bf a1 = frag_lds(As[cur], 32 * mg + 16, kk);
      v16bf b  = frag_lds(Ws[cur], nt * 16, kk);
      acc0 = __builtin_amdgcn_wmma_f32_16x16x32_bf16(false, a0, false, b, (short)0, acc0, false, false);
      acc1 = __builtin_amdgcn_wmma_f32_16x16x32_bf16(false, a1, false, b, (short)0, acc1, false, false);
    }
    async_join();
  }

  const int n = ocol0 + nt * 16 + (lane & 15);
  const float bpv = bp[n];
  const int hiOff = (lane >> 4) * 8;
#pragma unroll
  for (int r = 0; r < 8; ++r) {
    p[(size_t)(32 * mg + hiOff + r) * On + n] = acc0[r] + bpv;
    p[(size_t)(32 * mg + 16 + hiOff + r) * On + n] = acc1[r] + bpv;
  }
}

// ---------------------------------------------------------------------------
// Row softmax over O=1024 (one block per row)
// ---------------------------------------------------------------------------
__global__ __launch_bounds__(256)
void softmax_rows(const float* __restrict__ p, float* __restrict__ out) {
  __shared__ float red[256];
  const int row = blockIdx.x, tid = threadIdx.x;
  const float* pr = p + (size_t)row * On;

  float mx = -INFINITY;
  for (int j = tid; j < On; j += 256) mx = fmaxf(mx, pr[j]);
  red[tid] = mx; __syncthreads();
  for (int s = 128; s > 0; s >>= 1) {
    if (tid < s) red[tid] = fmaxf(red[tid], red[tid + s]);
    __syncthreads();
  }
  mx = red[0]; __syncthreads();

  float sum = 0.0f;
  for (int j = tid; j < On; j += 256) sum += __expf(pr[j] - mx);
  red[tid] = sum; __syncthreads();
  for (int s = 128; s > 0; s >>= 1) {
    if (tid < s) red[tid] += red[tid + s];
    __syncthreads();
  }
  const float inv = 1.0f / red[0];
  for (int j = tid; j < On; j += 256)
    out[(size_t)row * On + j] = __expf(pr[j] - mx) * inv;
}

// ---------------------------------------------------------------------------
// Host launcher
// ---------------------------------------------------------------------------
extern "C" void kernel_launch(void* const* d_in, const int* in_sizes, int n_in,
                              void* d_out, int out_size, void* d_ws, size_t ws_size,
                              hipStream_t stream) {
  const float* x  = (const float*)d_in[0];
  const float* Wx = (const float*)d_in[1];
  const float* bx = (const float*)d_in[2];
  const float* Wh = (const float*)d_in[3];
  const float* Wp = (const float*)d_in[4];
  const float* bp = (const float*)d_in[5];

  char* ws = (char*)d_ws;
  unsigned short* xb   = (unsigned short*)(ws);             // 8 MiB  : x bf16 [B,T,D]
  unsigned short* Wcat = (unsigned short*)(ws + 8388608);   // 10 MiB : [4H,1280] bf16
  unsigned short* Wpb  = (unsigned short*)(ws + 18874368);  // 2 MiB  : Wp bf16
  unsigned short* h0   = (unsigned short*)(ws + 20971520);  // 256 KiB: h ping
  unsigned short* h1   = (unsigned short*)(ws + 21233664);  // 256 KiB: h pong
  float* c             = (float*)(ws + 21495808);           // 512 KiB: cell state
  float* p             = (float*)(ws + 22020096);           // 512 KiB: logits

  conv_f32_bf16<<<(Bn * Tn * Dn + 255) / 256, 256, 0, stream>>>(x, xb, Bn * Tn * Dn);
  build_wcat<<<(4 * Hn * KCAT + 255) / 256, 256, 0, stream>>>(Wx, Wh, Wcat);
  conv_f32_bf16<<<(On * Hn + 255) / 256, 256, 0, stream>>>(Wp, Wpb, On * Hn);
  init_state<<<(Bn * Hn + 255) / 256, 256, 0, stream>>>(h0, c, Bn * Hn);

  unsigned short* hin = h0;
  unsigned short* hout = h1;
  for (int t = 0; t < Tn; ++t) {
    lstm_step<<<Hn / 32, 256, 0, stream>>>(xb, Wcat, bx, hin, hout, c, t);
    unsigned short* tmp = hin; hin = hout; hout = tmp;
  }

  proj_kernel<<<On / 32, 256, 0, stream>>>(hin, Wpb, bp, p);
  softmax_rows<<<Bn, 256, 0, stream>>>(p, (float*)d_out);
}
